// TESTAM_73770358276547
// MI455X (gfx1250) — compile-verified, matrix operand order checked by reading, not run on hardware
//
#include <hip/hip_runtime.h>
#include <hip/hip_bf16.h>
#include <math.h>

typedef __attribute__((ext_vector_type(16))) _Float16 v16h;
typedef __attribute__((ext_vector_type(8)))  _Float16 v8h;
typedef __attribute__((ext_vector_type(8)))  float    v8f;

#define B_    64
#define N_    325
#define T_    12
#define HID_  64
#define MH_   32
#define MS_   20
#define E_    3
#define BN_   (B_ * N_)
#define ITERS 8

// ---- 16-lane rotate-butterfly via DPP16 ROW_ROR (pure VALU, no DS/waits) ----
template <int K>
__device__ __forceinline__ float ror16(float x) {
  return __int_as_float(__builtin_amdgcn_update_dpp(
      0, __float_as_int(x), 0x120 + K, 0xF, 0xF, true));
}
__device__ __forceinline__ float red_sum16(float v) {
  v += ror16<1>(v);
  v += ror16<2>(v);
  v += ror16<4>(v);
  v += ror16<8>(v);
  return v;
}

// fast hardware reciprocal (v_rcp_f32, ~1 ulp): plenty for softmax/cosine scaling
__device__ __forceinline__ float frcp(float x) { return __builtin_amdgcn_rcpf(x); }

__device__ __forceinline__ v8f wmma32(v16h a, v16h b, v8f c) {
  // D = A(16x32 f16) * B(32x16 f16) + C(16x16 f32)
  return __builtin_amdgcn_wmma_f32_16x16x32_f16(false, a, false, b, (short)0, c,
                                                false, false);
}

// per-lane K position inside a 16x32 f16 A-operand / 32x16 B-operand (ISA layout)
__device__ __forceinline__ int kposf(int hi, int j) {
  return (j < 8) ? (hi * 8 + j) : (16 + hi * 8 + (j - 8));
}

// A operand (16x32 f16) from global h[t][k]
__device__ __forceinline__ v16h load_h_tile(const float* __restrict__ hb, int c) {
  const int lane = threadIdx.x & 31;
  const int t = lane & 15;
  const int hi = lane >> 4;
  v16h a;
  if (t < T_) {
    const float* p = hb + t * HID_ + c * 32 + hi * 8;
#pragma unroll
    for (int j = 0; j < 8; ++j) {
      a[j]     = (_Float16)p[j];
      a[8 + j] = (_Float16)p[16 + j];
    }
  } else {
#pragma unroll
    for (int j = 0; j < 16; ++j) a[j] = (_Float16)0.f;
  }
  return a;
}

// B operand (32x16 f16) gathered from a row-major [64][32] f32 weight matrix
__device__ __forceinline__ v16h load_w_tile(const float* __restrict__ W, int kc, int nc) {
  const int lane = threadIdx.x & 31;
  const int n = nc * 16 + (lane & 15);
  const int hi = lane >> 4;
  v16h w;
#pragma unroll
  for (int j = 0; j < 16; ++j)
    w[j] = (_Float16)W[(kc * 32 + kposf(hi, j)) * MH_ + n];
  return w;
}

// Operand fetch from row-major [rows][32] f16 LDS tile (two contiguous 8-half runs)
__device__ __forceinline__ v16h lds_tile(const _Float16* base, int rowOff) {
  const int lane = threadIdx.x & 31;
  const int r = rowOff + (lane & 15);
  const int hi = lane >> 4;
  const _Float16* p = base + r * 32 + hi * 8;
  v16h v;
#pragma unroll
  for (int j = 0; j < 8; ++j) { v[j] = p[j]; v[8 + j] = p[16 + j]; }
  return v;
}

// Operand tile staged lane-major in LDS: lane's 16 halfs contiguous (2x ds_load_b128)
__device__ __forceinline__ v16h lds_wtile(const _Float16* tileBase) {
  return *(const v16h*)(tileBase + (threadIdx.x & 31) * 16);
}

__global__ __launch_bounds__(128) __attribute__((amdgpu_waves_per_eu(4)))
void testam_kernel(
    const float* __restrict__ input, const float* __restrict__ h0,
    const float* __restrict__ h1, const float* __restrict__ h2,
    const float* __restrict__ memory, const float* __restrict__ input_query,
    const float* __restrict__ hid_query, const float* __restrict__ key_w,
    const float* __restrict__ value_w, float* __restrict__ out) {
  // operand staging: weights pre-swizzled per lane (12 tiles/expert, 4 for wave 3)
  __shared__ __align__(32) _Float16 sW[E_][12][32 * 16];   // 36 KB
  __shared__ __align__(32) _Float16 sW3[4][32 * 16];       // 4 KB
  __shared__ __align__(16) _Float16 sQ[E_][16 * 32];
  __shared__ __align__(16) _Float16 sK[E_][16 * 32];
  __shared__ __align__(16) _Float16 sA[E_][16 * 32];       // attn, K-padded zeros
  __shared__ __align__(16) _Float16 sVT[E_][32 * 32];      // V transposed [mh][t]
  __shared__ __align__(16) _Float16 sP[16 * 32];           // mem_score (wave 3)
  __shared__ float sMem[16 * 32];
  __shared__ float sNorm[16];

  const int tid = threadIdx.x;
  const int wid = tid >> 5;
  const int lane = tid & 31;
  const int s = lane & 15;
  const int hi = lane >> 4;

  // zero LDS pad regions once (WMMA operands must see 0, not junk/NaN)
  for (int i = tid; i < E_ * 16 * 32; i += 128) {
    ((_Float16*)sQ)[i] = (_Float16)0.f;
    ((_Float16*)sK)[i] = (_Float16)0.f;
    ((_Float16*)sA)[i] = (_Float16)0.f;
  }
  for (int i = tid; i < E_ * 32 * 32; i += 128) ((_Float16*)sVT)[i] = (_Float16)0.f;
  for (int i = tid; i < 16 * 32; i += 128) { sMem[i] = 0.f; sP[i] = (_Float16)0.f; }
  for (int i = tid; i < 16; i += 128) sNorm[i] = 0.f;

  // ---- stage operand tiles into LDS (weights are L2-hot; f16 convert once) ----
  v16h iqh0, iqh1, iqh2;   // wave 3: input_query gathered per lane (f16)
  const float* hsrc = nullptr;

  if (wid < E_) {
    const float* Wm[3] = {hid_query + wid * HID_ * MH_, key_w + wid * HID_ * MH_,
                          value_w + wid * HID_ * MH_};
#pragma unroll
    for (int m = 0; m < 3; ++m)
#pragma unroll
      for (int kc = 0; kc < 2; ++kc)
#pragma unroll
        for (int nc = 0; nc < 2; ++nc) {
          v16h w = load_w_tile(Wm[m], kc, nc);
          *(v16h*)&sW[wid][m * 4 + kc * 2 + nc][lane * 16] = w;
        }
    hsrc = (wid == 0) ? h0 : (wid == 1) ? h1 : h2;
  } else {
    v16h bm0, bm1, bq0, bq1;
#pragma unroll
    for (int j = 0; j < 16; ++j) {
      const int k = kposf(hi, j);
      iqh0[j] = (_Float16)input_query[0 * MH_ + k];
      iqh1[j] = (_Float16)input_query[1 * MH_ + k];
      iqh2[j] = (_Float16)input_query[2 * MH_ + k];
      // scores B: B[k=mh][n=slot] = memory[slot][mh]; slots >=20 zero
      bm0[j] = (_Float16)memory[s * MH_ + k];
      bm1[j] = (16 + s < MS_) ? (_Float16)memory[(16 + s) * MH_ + k]
                              : (_Float16)0.f;
      // memories B: B[k=slot][n=mh] = memory[slot][mh]; slots >=20 zero
      bq0[j] = (k < MS_) ? (_Float16)memory[k * MH_ + s] : (_Float16)0.f;
      bq1[j] = (k < MS_) ? (_Float16)memory[k * MH_ + 16 + s] : (_Float16)0.f;
    }
    *(v16h*)&sW3[0][lane * 16] = bm0;
    *(v16h*)&sW3[1][lane * 16] = bm1;
    *(v16h*)&sW3[2][lane * 16] = bq0;
    *(v16h*)&sW3[3][lane * 16] = bq1;
  }
  __syncthreads();

  const int bn0 = blockIdx.x * ITERS;
  const v8f z8 = {};
  v8f ha0 = z8, ha1 = z8;

#pragma unroll 1
  for (int it = 0; it < ITERS; ++it) {
    const int bn = bn0 + it;
    const bool valid = bn < BN_;

    if (valid && wid < E_) {
      const float* hb = hsrc + (size_t)bn * T_ * HID_;
      if (bn + 1 < BN_)
        __builtin_prefetch(hsrc + (size_t)(bn + 1) * T_ * HID_ + lane * 16, 0, 1);

      // ---- QKV projections: 12 WMMAs, weights streamed from LDS ----
      v16h a0 = load_h_tile(hb, 0);
      v16h a1 = load_h_tile(hb, 1);
      v8f qa0 = z8, qa1 = z8, ka0 = z8, ka1 = z8, va0 = z8, va1 = z8;
      qa0 = wmma32(a0, lds_wtile(sW[wid][0]), qa0);
      qa0 = wmma32(a1, lds_wtile(sW[wid][2]), qa0);
      qa1 = wmma32(a0, lds_wtile(sW[wid][1]), qa1);
      qa1 = wmma32(a1, lds_wtile(sW[wid][3]), qa1);
      ka0 = wmma32(a0, lds_wtile(sW[wid][4]), ka0);
      ka0 = wmma32(a1, lds_wtile(sW[wid][6]), ka0);
      ka1 = wmma32(a0, lds_wtile(sW[wid][5]), ka1);
      ka1 = wmma32(a1, lds_wtile(sW[wid][7]), ka1);
      va0 = wmma32(a0, lds_wtile(sW[wid][8]), va0);
      va0 = wmma32(a1, lds_wtile(sW[wid][10]), va0);
      va1 = wmma32(a0, lds_wtile(sW[wid][9]), va1);
      va1 = wmma32(a1, lds_wtile(sW[wid][11]), va1);

      // spill Q,K row-major (per-lane column); V transposed packed as b128
      v8h pv0, pv1;
#pragma unroll
      for (int g = 0; g < 8; ++g) {
        const int t = g + 8 * hi;
        sQ[wid][t * 32 + s]      = (_Float16)qa0[g];
        sQ[wid][t * 32 + 16 + s] = (_Float16)qa1[g];
        sK[wid][t * 32 + s]      = (_Float16)ka0[g];
        sK[wid][t * 32 + 16 + s] = (_Float16)ka1[g];
        pv0[g] = (_Float16)va0[g];
        pv1[g] = (_Float16)va1[g];
      }
      *(v8h*)&sVT[wid][s * 32 + 8 * hi]        = pv0;
      *(v8h*)&sVT[wid][(16 + s) * 32 + 8 * hi] = pv1;
      asm volatile("s_wait_dscnt 0" ::: "memory");

      // ---- energy = Q K^T ----
      v16h aq = lds_tile(sQ[wid], 0);
      v16h bk = lds_tile(sK[wid], 0);
      v8f en = z8;
      en = wmma32(aq, bk, en);

      // ---- row softmax over s; energies are O(10), so no max-shift needed;
      //      masked pad cols use exp(-3e38) == 0 ----
#pragma unroll
      for (int g = 0; g < 8; ++g) {
        float v = (s < T_) ? en[g] : -3.0e38f;
        float ex = __expf(v);
        float sm = red_sum16(ex);
        sA[wid][(g + 8 * hi) * 32 + s] = (_Float16)(ex * frcp(sm));
      }
      asm volatile("s_wait_dscnt 0" ::: "memory");

      // ---- hidden_att = attn V ----
      v16h aat = lds_tile(sA[wid], 0);
      v16h bv0 = lds_tile(sVT[wid], 0);
      v16h bv1 = lds_tile(sVT[wid], 16);
      ha0 = z8; ha1 = z8;
      ha0 = wmma32(aat, bv0, ha0);
      ha1 = wmma32(aat, bv1, ha1);
    } else if (valid) {
      // ==== wave 3: memory path as 4 WMMAs ====
      float in0 = 0.f, in1 = 0.f, in2 = 0.f;
      if (s < T_) {
        const float* ip = input + ((size_t)bn * T_ + s) * 3;
        in0 = ip[0]; in1 = ip[1]; in2 = ip[2];
      }
      // A operand q[t][mh]: K=3 contraction as packed f16 FMAs
      v16h aq = iqh0 * (_Float16)in0;
      aq += iqh1 * (_Float16)in1;
      aq += iqh2 * (_Float16)in2;

      v8f e0 = z8, e1 = z8;
      e0 = wmma32(aq, lds_wtile(sW3[0]), e0);
      e1 = wmma32(aq, lds_wtile(sW3[1]), e1);

      // softmax over 20 slots (cols 0..15 in e0, 16..19 in e1); scores are tiny,
      // no max-shift needed; pad slots underflow to 0
#pragma unroll
      for (int g = 0; g < 8; ++g) {
        float v1 = (s < MS_ - 16) ? e1[g] : -3.0e38f;
        float x0 = __expf(e0[g]);
        float x1 = __expf(v1);
        float sm = red_sum16(x0) + red_sum16(x1);
        const float inv = frcp(sm);
        sP[(g + 8 * hi) * 32 + s]      = (_Float16)(x0 * inv);
        sP[(g + 8 * hi) * 32 + 16 + s] = (_Float16)(x1 * inv);
      }
      asm volatile("s_wait_dscnt 0" ::: "memory");

      v16h aP = lds_tile(sP, 0);
      v8f m0 = z8, m1 = z8;
      m0 = wmma32(aP, lds_wtile(sW3[2]), m0);
      m1 = wmma32(aP, lds_wtile(sW3[3]), m1);

#pragma unroll
      for (int g = 0; g < 8; ++g) {
        const int t = g + 8 * hi;
        sMem[t * 32 + s]      = m0[g];
        sMem[t * 32 + 16 + s] = m1[g];
        float ssq = red_sum16(m0[g] * m0[g] + m1[g] * m1[g]);
        if (s == 0) sNorm[t] = sqrtf(ssq);
      }
    }

    __syncthreads();  // memories published; previous iteration fully consumed

    if (valid && wid < E_) {
      // ---- cosine(memories, hidden_att) ----
#pragma unroll
      for (int g = 0; g < 8; ++g) {
        const int t = g + 8 * hi;
        const float m0 = sMem[t * 32 + s];
        const float m1 = sMem[t * 32 + 16 + s];
        float dp = ha0[g] * m0 + ha1[g] * m1;
        float ss = ha0[g] * ha0[g] + ha1[g] * ha1[g];
        dp = red_sum16(dp);
        ss = red_sum16(ss);
        if (s == 0 && t < T_) {
          const float denom = fmaxf(sNorm[t] * sqrtf(ss), 1e-8f);
          out[((size_t)bn * T_ + t) * E_ + wid] = dp * frcp(denom);
        }
      }
    }
    __syncthreads();  // cosine reads done before wave3 overwrites sMem
  }
}

extern "C" void kernel_launch(void* const* d_in, const int* in_sizes, int n_in,
                              void* d_out, int out_size, void* d_ws, size_t ws_size,
                              hipStream_t stream) {
  (void)in_sizes; (void)n_in; (void)out_size; (void)d_ws; (void)ws_size;
  const float* input       = (const float*)d_in[0];
  const float* h0          = (const float*)d_in[1];
  const float* h1          = (const float*)d_in[2];
  const float* h2          = (const float*)d_in[3];
  const float* memory      = (const float*)d_in[4];
  const float* input_query = (const float*)d_in[5];
  const float* hid_query   = (const float*)d_in[6];
  const float* key_w       = (const float*)d_in[7];
  const float* value_w     = (const float*)d_in[8];
  float* out = (float*)d_out;

  const int blocks = (BN_ + ITERS - 1) / ITERS;  // 2600
  hipLaunchKernelGGL(testam_kernel, dim3(blocks), dim3(128), 0, stream,
                     input, h0, h1, h2, memory, input_query, hid_query, key_w,
                     value_w, out);
}
